// Decoder_13589276524762
// MI455X (gfx1250) — matrix operand order repaired and checked by reading, not compiled, not run on hardware
//
#include <hip/hip_runtime.h>

// ---------------------------------------------------------------------------
// CDNA5 (gfx1250) fused decoder: gather+concat -> bf16 WMMA GEMM -> BN -> LReLU
// wave32, V_WMMA_F32_16X16X32_BF16, deterministic BN reductions.
// ---------------------------------------------------------------------------

typedef __attribute__((ext_vector_type(16))) __bf16     v16bf;
typedef __attribute__((ext_vector_type(8)))  float      v8f;
typedef __attribute__((ext_vector_type(8)))  unsigned int u32x8;
typedef __attribute__((ext_vector_type(4)))  unsigned int u32x4;

__device__ __forceinline__ unsigned short f2bf(float f) {
  union { float f; unsigned int u; } v; v.f = f;
  unsigned int u = v.u;
  return (unsigned short)((u + 0x7FFFu + ((u >> 16) & 1u)) >> 16); // RNE
}

// ---------------------------------------------------------------------------
// Pack f32 weights [K][N] into per-lane WMMA B-fragment layout, bf16, padded.
// Layout: Wp[((c*NT + t)*32 + lane)*16 + j] where the lane's 16 bf16 values
// are exactly its V_WMMA B operand for K-chunk c, N-tile t:
//   k = c*32 + (lane>=16 ? 16 : 0) + j,  n = t*16 + (lane & 15)
// ---------------------------------------------------------------------------
__global__ __launch_bounds__(256) void pack_w_kernel(
    const float* __restrict__ W, int K, int N, int NT,
    unsigned short* __restrict__ Wp, int total)
{
  int tid = blockIdx.x * 256 + threadIdx.x;
  if (tid >= total) return;
  int j    = tid & 15;
  int lane = (tid >> 4) & 31;
  int ct   = tid >> 9;
  int t    = ct % NT;
  int c    = ct / NT;
  int k = c * 32 + ((lane >= 16) ? 16 : 0) + j;
  int n = t * 16 + (lane & 15);
  float v = (k < K && n < N) ? W[(long)k * N + n] : 0.0f;
  Wp[tid] = f2bf(v);
}

// ---------------------------------------------------------------------------
// Fused gather+concat+GEMM (+bias) with per-block deterministic BN partials.
// Block = 256 threads = 8 wave32, computes 128 rows x (NT*16) cols.
// A tile (128x32 bf16) staged in LDS from gathered prev rows + skip rows.
// ---------------------------------------------------------------------------
template <int NT, bool BN>
__global__ __launch_bounds__(256) void fused_gemm_kernel(
    const float* __restrict__ prev, int prevStride, int Kg,
    const int*   __restrict__ idx,
    const float* __restrict__ skip, int Ks,
    const unsigned short* __restrict__ Wp, int KC,
    const float* __restrict__ bias,
    float* __restrict__ out, int outStride, int outCols,
    float* __restrict__ partS, float* __restrict__ partQ)
{
  __shared__ unsigned short As[128 * 32];   // 8 KB bf16 A tile
  __shared__ float pS[NT * 16][16];
  __shared__ float pQ[NT * 16][16];

  const int tid  = threadIdx.x;
  const int lane = tid & 31;
  const int wave = tid >> 5;
  const long mBase = (long)blockIdx.x * 128;

  // A staging: each thread feeds 16 contiguous k of one row
  const int  ldRow = tid >> 1;
  const int  ldK   = (tid & 1) << 4;
  const long gRow  = mBase + ldRow;
  const float* prow = prev + (long)idx[gRow] * prevStride;
  const float* srow = skip + gRow * (long)Ks;

  v8f acc[NT];
#pragma unroll
  for (int t = 0; t < NT; ++t)
#pragma unroll
    for (int v = 0; v < 8; ++v) acc[t][v] = 0.0f;

  // A-fragment read position (contiguous k order, matches packed B):
  // lane<16: K 0..15 ; lane>=16: K 16..31, row = lane%16
  const int fRow  = (wave << 4) + (lane & 15);
  const int fHalf = (lane < 16) ? 0 : 16;          // in ushorts
  const unsigned short* aRow = &As[fRow * 32];

  const u32x8* wp = (const u32x8*)Wp;              // 32B per lane-fragment

  for (int c = 0; c < KC; ++c) {
    __syncthreads();
    { // gather + concat + f32->bf16 into LDS
      const int k0 = c * 32 + ldK;
      unsigned int pk[8];
#pragma unroll
      for (int h = 0; h < 8; ++h) {
        int ka = k0 + 2 * h, kb = k0 + 2 * h + 1;
        float va = (ka < Kg) ? prow[ka] : ((ka < Kg + Ks) ? srow[ka - Kg] : 0.0f);
        float vb = (kb < Kg) ? prow[kb] : ((kb < Kg + Ks) ? srow[kb - Kg] : 0.0f);
        pk[h] = (unsigned int)f2bf(va) | ((unsigned int)f2bf(vb) << 16);
      }
      u32x4* dst = (u32x4*)&As[ldRow * 32 + ldK];
      u32x4 w0, w1;
      w0.x = pk[0]; w0.y = pk[1]; w0.z = pk[2]; w0.w = pk[3];
      w1.x = pk[4]; w1.y = pk[5]; w1.z = pk[6]; w1.w = pk[7];
      dst[0] = w0; dst[1] = w1;
    }
    __syncthreads();

    // load A fragment (two ds_load_b128)
    u32x4 a0 = *(const u32x4*)(aRow + fHalf);
    u32x4 a1 = *(const u32x4*)(aRow + fHalf + 8);
    u32x8 au;
    au[0] = a0.x; au[1] = a0.y; au[2] = a0.z; au[3] = a0.w;
    au[4] = a1.x; au[5] = a1.y; au[6] = a1.z; au[7] = a1.w;
    v16bf a = __builtin_bit_cast(v16bf, au);

#pragma unroll
    for (int t = 0; t < NT; ++t) {
      u32x8 bu = wp[(size_t)(c * NT + t) * 32 + lane];
      v16bf b = __builtin_bit_cast(v16bf, bu);
      acc[t] = __builtin_amdgcn_wmma_f32_16x16x32_bf16(
          false, a, false, b, (short)0, acc[t], false, false);
    }
  }

  // epilogue: bias add, store, BN partial sums (fixed-order -> deterministic)
  const long rBase = mBase + (wave << 4) + ((lane < 16) ? 0 : 8);
  const int  slot  = (wave << 1) + (lane >> 4);   // 0..15, unique per (col,block)
#pragma unroll
  for (int t = 0; t < NT; ++t) {
    const int  col  = t * 16 + (lane & 15);
    const bool live = (col < outCols);
    const float bv  = live ? bias[col] : 0.0f;
    float s = 0.0f, q = 0.0f;
#pragma unroll
    for (int v = 0; v < 8; ++v) {
      float y = acc[t][v] + bv;
      if (live) out[(rBase + v) * (long)outStride + col] = y;
      s += y; q += y * y;
    }
    if (BN) { pS[col][slot] = s; pQ[col][slot] = q; }
  }
  if (BN) {
    __syncthreads();
    for (int c2 = tid; c2 < NT * 16; c2 += 256) {
      float s = 0.0f, q = 0.0f;
#pragma unroll
      for (int k = 0; k < 16; ++k) { s += pS[c2][k]; q += pQ[c2][k]; }
      partS[(size_t)blockIdx.x * (NT * 16) + c2] = s;
      partQ[(size_t)blockIdx.x * (NT * 16) + c2] = q;
    }
  }
}

// Deterministic per-column reduction of block partials
__global__ __launch_bounds__(256) void reduce_cols_kernel(
    const float* __restrict__ partS, const float* __restrict__ partQ,
    int nb, int C, float* __restrict__ outS, float* __restrict__ outQ)
{
  int c = blockIdx.x * 256 + threadIdx.x;
  if (c >= C) return;
  float s = 0.0f, q = 0.0f;
  for (int b = 0; b < nb; ++b) {
    s += partS[(size_t)b * C + c];
    q += partQ[(size_t)b * C + c];
  }
  outS[c] = s; outQ[c] = q;
}

// Training-mode BN (biased var) + LeakyReLU, in place on compact [M x C]
__global__ __launch_bounds__(256) void bn_lrelu_kernel(
    float* __restrict__ Y, float invM, int C, long total,
    const float* __restrict__ colS, const float* __restrict__ colQ,
    const float* __restrict__ g, const float* __restrict__ be)
{
  long i = (long)blockIdx.x * 256 + threadIdx.x;
  if (i >= total) return;
  int c = (int)(i % C);
  float m   = colS[c] * invM;
  float var = colQ[c] * invM - m * m;
  float sc  = rsqrtf(var + 1e-5f) * g[c];
  float y   = (Y[i] - m) * sc + be[c];
  Y[i] = (y >= 0.0f) ? y : 0.01f * y;
}

// ---------------------------------------------------------------------------
extern "C" void kernel_launch(void* const* d_in, const int* in_sizes, int n_in,
                              void* d_out, int out_size, void* d_ws, size_t ws_size,
                              hipStream_t stream)
{
  (void)in_sizes; (void)n_in; (void)out_size; (void)ws_size;
  const float* feats = (const float*)d_in[0];
  const float* skip1 = (const float*)d_in[1];
  const float* skip2 = (const float*)d_in[2];
  const float* skip3 = (const float*)d_in[3];
  const int*   idx1  = (const int*)d_in[4];
  const int*   idx2  = (const int*)d_in[5];
  const int*   idx3  = (const int*)d_in[6];
  const float* W1  = (const float*)d_in[7];
  const float* b1  = (const float*)d_in[8];
  const float* g1  = (const float*)d_in[9];
  const float* be1 = (const float*)d_in[10];
  const float* W2  = (const float*)d_in[11];
  const float* b2  = (const float*)d_in[12];
  const float* g2  = (const float*)d_in[13];
  const float* be2 = (const float*)d_in[14];
  const float* W3  = (const float*)d_in[15];
  const float* b3  = (const float*)d_in[16];
  float* out = (float*)d_out;

  constexpr int N2 = 16384, N1 = 65536, N0 = 262144;
  // K chunks / N tiles per stage (K padded to x32, N padded to x16)
  constexpr int KC1 = 25, NT1 = 9;   // K 770->800, N 129->144
  constexpr int KC2 = 13, NT2 = 4;   // K 385->416, N 64
  constexpr int KC3 = 6,  NT3 = 3;   // K 192,      N 34->48

  // workspace carving (256B aligned regions)
  char* p = (char*)d_ws;
  auto carve = [&](size_t bytes) -> void* {
    void* r = (void*)p;
    p += (bytes + 255) & ~(size_t)255;
    return r;
  };
  unsigned short* Wp1 = (unsigned short*)carve((size_t)KC1 * NT1 * 512 * 2);
  unsigned short* Wp2 = (unsigned short*)carve((size_t)KC2 * NT2 * 512 * 2);
  unsigned short* Wp3 = (unsigned short*)carve((size_t)KC3 * NT3 * 512 * 2);
  float* Y1    = (float*)carve((size_t)N2 * 129 * 4);
  float* Y2    = (float*)carve((size_t)N1 * 64 * 4);
  float* p1S   = (float*)carve((size_t)128 * NT1 * 16 * 4);
  float* p1Q   = (float*)carve((size_t)128 * NT1 * 16 * 4);
  float* p2S   = (float*)carve((size_t)512 * NT2 * 16 * 4);
  float* p2Q   = (float*)carve((size_t)512 * NT2 * 16 * 4);
  float* sum1  = (float*)carve((size_t)NT1 * 16 * 4);
  float* sq1   = (float*)carve((size_t)NT1 * 16 * 4);
  float* sum2  = (float*)carve((size_t)NT2 * 16 * 4);
  float* sq2   = (float*)carve((size_t)NT2 * 16 * 4);

  // ---- weight packing (bf16, WMMA B-fragment layout) ----
  {
    int t1 = KC1 * NT1 * 512, t2 = KC2 * NT2 * 512, t3 = KC3 * NT3 * 512;
    pack_w_kernel<<<(t1 + 255) / 256, 256, 0, stream>>>(W1, 770, 129, NT1, Wp1, t1);
    pack_w_kernel<<<(t2 + 255) / 256, 256, 0, stream>>>(W2, 385, 64,  NT2, Wp2, t2);
    pack_w_kernel<<<(t3 + 255) / 256, 256, 0, stream>>>(W3, 192, 34,  NT3, Wp3, t3);
  }

  // ---- stage 1: [N2,770] @ W1 -> [N2,129], BN + LReLU ----
  fused_gemm_kernel<NT1, true><<<N2 / 128, 256, 0, stream>>>(
      feats, 258, 258, idx1, skip3, 512, Wp1, KC1, b1,
      Y1, 129, 129, p1S, p1Q);
  reduce_cols_kernel<<<1, 256, 0, stream>>>(p1S, p1Q, 128, NT1 * 16, sum1, sq1);
  {
    long tot = (long)N2 * 129;
    bn_lrelu_kernel<<<(int)((tot + 255) / 256), 256, 0, stream>>>(
        Y1, 1.0f / (float)N2, 129, tot, sum1, sq1, g1, be1);
  }

  // ---- stage 2: [N1,385] @ W2 -> [N1,64], BN + LReLU ----
  fused_gemm_kernel<NT2, true><<<N1 / 128, 256, 0, stream>>>(
      Y1, 129, 129, idx2, skip2, 256, Wp2, KC2, b2,
      Y2, 64, 64, p2S, p2Q);
  reduce_cols_kernel<<<1, 256, 0, stream>>>(p2S, p2Q, 512, NT2 * 16, sum2, sq2);
  {
    long tot = (long)N1 * 64;
    bn_lrelu_kernel<<<(int)((tot + 255) / 256), 256, 0, stream>>>(
        Y2, 1.0f / (float)N1, 64, tot, sum2, sq2, g2, be2);
  }

  // ---- stage 3: [N0,192] @ W3 + b3 -> [N0,34] (no BN / activation) ----
  fused_gemm_kernel<NT3, false><<<N0 / 128, 256, 0, stream>>>(
      Y2, 64, 64, idx3, skip1, 128, Wp3, KC3, b3,
      out, 34, 34, nullptr, nullptr);
}
